// AttnActGPT2Attention_2345052144048
// MI455X (gfx1250) — matrix operand order repaired and checked by reading, not compiled, not run on hardware
//
#include <hip/hip_runtime.h>
#include <hip/hip_bf16.h>
#include <stdint.h>

typedef __attribute__((ext_vector_type(16))) __bf16 v16bf;
typedef __attribute__((ext_vector_type(8)))  __bf16 v8bf;
typedef __attribute__((ext_vector_type(8)))  float  v8f;

#define NEG_INF (-__builtin_inff())

// ---------------------------------------------------------------------------
// WMMA helpers (CDNA5 v_wmma_f32_16x16x32_bf16, wave32)
// A-fragment (16-bit A 16x32, contiguous k order used here): lane L<16 -> row M=L,
//   elems 0..15 = K 0..15 ; lanes 16..31 -> K+16.
// B-fragment: lane L<16 -> col N=L, elems 0..15 = K 0..15 ; lanes 16..31 K+16.
// C/D: vgpr r, lanes<16 -> M=r, N=lane ; lanes>=16 -> M=8+r, N=lane-16.
// ---------------------------------------------------------------------------
__device__ __forceinline__ v16bf frag_a(const __bf16* row, int kb) {
  const v8bf lo = *(const v8bf*)(row + kb);
  const v8bf hi = *(const v8bf*)(row + kb + 8);
  v16bf r;
#pragma unroll
  for (int i = 0; i < 8; ++i) { r[i] = lo[i]; r[i + 8] = hi[i]; }
  return r;
}

__device__ __forceinline__ v16bf frag_b(const __bf16* col, int kb) {
  const v8bf lo = *(const v8bf*)(col + kb);
  const v8bf hi = *(const v8bf*)(col + kb + 8);
  v16bf r;
#pragma unroll
  for (int i = 0; i < 8; ++i) { r[i] = lo[i]; r[i + 8] = hi[i]; }
  return r;
}

__device__ __forceinline__ v8f wmma_bf16(v16bf a, v16bf b, v8f c) {
  return __builtin_amdgcn_wmma_f32_16x16x32_bf16(
      /*neg_a=*/false, a, /*neg_b=*/false, b,
      /*c_mod=*/(short)0, c, /*reuse_a=*/false, /*reuse_b=*/false);
}

// ---------------------------------------------------------------------------
// Prepass 1: elementwise fp32 -> bf16 (n multiple of 8)
// ---------------------------------------------------------------------------
__global__ __launch_bounds__(256) void convert_f32_bf16(
    const float* __restrict__ src, __bf16* __restrict__ dst, long n) {
  const long i = ((long)blockIdx.x * 256 + threadIdx.x) * 8;
  if (i >= n) return;
  const float4 a = *(const float4*)(src + i);
  const float4 b = *(const float4*)(src + i + 4);
  v8bf o;
  o[0] = (__bf16)a.x; o[1] = (__bf16)a.y; o[2] = (__bf16)a.z; o[3] = (__bf16)a.w;
  o[4] = (__bf16)b.x; o[5] = (__bf16)b.y; o[6] = (__bf16)b.z; o[7] = (__bf16)b.w;
  *(v8bf*)(dst + i) = o;
}

// ---------------------------------------------------------------------------
// Prepass 2: src[K,N] fp32  ->  dst[N,K] bf16  (32x32 LDS tile transpose)
// ---------------------------------------------------------------------------
__global__ __launch_bounds__(256) void transpose_convert(
    const float* __restrict__ src, __bf16* __restrict__ dst, int K, int N) {
  __shared__ float tile[32][33];
  const int n0 = blockIdx.x * 32, k0 = blockIdx.y * 32;
  const int tx = threadIdx.x & 31;
  const int ty = threadIdx.x >> 5;  // 0..7
#pragma unroll
  for (int i = 0; i < 4; ++i)
    tile[ty + i * 8][tx] = src[(size_t)(k0 + ty + i * 8) * N + n0 + tx];
  __syncthreads();
#pragma unroll
  for (int i = 0; i < 4; ++i)
    dst[(size_t)(n0 + ty + i * 8) * K + k0 + tx] = (__bf16)tile[tx][ty + i * 8];
}

// ---------------------------------------------------------------------------
// GEMM + bias, bf16 inputs: C[M,N] = A[M,K] * Wt[N,K]^T + bias[N]
// 256 threads = 8 waves; block tile 128x128, K-step 32.
// Wave grid 4(M) x 2(N); each wave computes 32x64 = 2x4 WMMA tiles
// (8 WMMAs per 12 ds_load_b128 per K-step). Tile staging is pure b128 copies.
// ---------------------------------------------------------------------------
#define G_BM 128
#define G_BN 128
#define G_BK 32
#define G_LD 48  // bf16 stride: 96B rows, fragment offsets stay 16B aligned

template <bool BF16_OUT>
__global__ __launch_bounds__(256) void gemm_bf16_bias(
    const __bf16* __restrict__ A,    // [M,K] bf16
    const __bf16* __restrict__ Wt,   // [N,K] bf16 (pre-transposed weights)
    const float* __restrict__ bias,  // [N] fp32
    void* __restrict__ Cv, int M, int N, int K) {
  __shared__ __bf16 As[G_BM * G_LD];
  __shared__ __bf16 Bs[G_BN * G_LD];

  const int tid  = threadIdx.x;
  const int lane = tid & 31;
  const int wave = tid >> 5;
  const int wm   = wave & 3;   // 0..3 (M)
  const int wn   = wave >> 2;  // 0..1 (N)
  const int half = lane >> 4;
  const int l16  = lane & 15;
  const int bm   = blockIdx.y * G_BM;
  const int bn   = blockIdx.x * G_BN;

  v8f acc[2][4];
#pragma unroll
  for (int m = 0; m < 2; ++m)
#pragma unroll
    for (int n = 0; n < 4; ++n)
#pragma unroll
      for (int r = 0; r < 8; ++r) acc[m][n][r] = 0.0f;

  for (int k0 = 0; k0 < K; k0 += G_BK) {
    // A tile: 128 x 32 bf16 = 512 x 16B chunks, 2 per thread
#pragma unroll
    for (int i = 0; i < 2; ++i) {
      const int idx = tid + i * 256;
      const int row = idx >> 2;
      const int c   = (idx & 3) * 8;
      *(uint4*)(As + row * G_LD + c) =
          *(const uint4*)(A + (size_t)(bm + row) * K + k0 + c);
    }
    // B tile from Wt: 128 x 32 bf16, same pattern (already K-contiguous)
#pragma unroll
    for (int i = 0; i < 2; ++i) {
      const int idx = tid + i * 256;
      const int row = idx >> 2;
      const int c   = (idx & 3) * 8;
      *(uint4*)(Bs + row * G_LD + c) =
          *(const uint4*)(Wt + (size_t)(bn + row) * K + k0 + c);
    }
    // prefetch next tiles into cache hierarchy
    if (k0 + G_BK < K) {
      __builtin_prefetch(A + (size_t)(bm + (tid >> 1)) * K + k0 + G_BK, 0, 0);
      __builtin_prefetch(Wt + (size_t)(bn + (tid >> 1)) * K + k0 + G_BK, 0, 0);
    }
    __syncthreads();

    v16bf af[2], bfr[4];
#pragma unroll
    for (int m = 0; m < 2; ++m)
      af[m] = frag_a(As + (wm * 32 + m * 16 + l16) * G_LD, half * 16);
#pragma unroll
    for (int n = 0; n < 4; ++n)
      bfr[n] = frag_b(Bs + (wn * 64 + n * 16 + l16) * G_LD, half * 16);
#pragma unroll
    for (int m = 0; m < 2; ++m)
#pragma unroll
      for (int n = 0; n < 4; ++n)
        acc[m][n] = wmma_bf16(af[m], bfr[n], acc[m][n]);
    __syncthreads();
  }

#pragma unroll
  for (int m = 0; m < 2; ++m)
#pragma unroll
    for (int n = 0; n < 4; ++n) {
      const int col = bn + wn * 64 + n * 16 + l16;
      const float bc = bias[col];
#pragma unroll
      for (int r = 0; r < 8; ++r) {
        const int row = bm + wm * 32 + m * 16 + half * 8 + r;
        const float val = acc[m][n][r] + bc;
        if constexpr (BF16_OUT)
          ((__bf16*)Cv)[(size_t)row * N + col] = (__bf16)val;
        else
          ((float*)Cv)[(size_t)row * N + col] = val;
      }
    }
}

// ---------------------------------------------------------------------------
// Flash-style causal attention, bf16 qkv input, bf16 output.
// Grid: (S/64, H, B). Block 128 threads = 4 waves; wave owns 16 query rows.
// Score scale 1/sqrt(Dh) folded in post-WMMA. K/V streamed per 64-key tile.
// ---------------------------------------------------------------------------
#define A_S  2048
#define A_D  1024
#define A_DH 64
#define A_BR 64
#define A_BC 64
#define A_LD 72  // 144B rows: fragment offsets stay 16B aligned

__global__ __launch_bounds__(128) void flash_attn_kernel(
    const __bf16* __restrict__ qkv, __bf16* __restrict__ out) {
  __shared__ __bf16 Qs[A_BR * A_LD];  // [q_row][dh]
  __shared__ __bf16 Ks[A_BC * A_LD];  // [key][dh]
  __shared__ __bf16 Vt[A_DH * A_LD];  // [dh][key]
  __shared__ __bf16 Ps[A_BR * A_LD];  // [q_row][key]

  const int tid  = threadIdx.x;
  const int lane = tid & 31;
  const int wave = tid >> 5;  // 0..3
  const int half = lane >> 4;
  const int l16  = lane & 15;
  const int qt   = blockIdx.x;
  const int h    = blockIdx.y;
  const int b    = blockIdx.z;

  const size_t rs     = 3 * A_D;
  const __bf16* qbase = qkv + (size_t)b * A_S * rs + h * A_DH;
  const __bf16* kbase = qbase + A_D;
  const __bf16* vbase = qbase + 2 * A_D;

  // Q tile: 64 rows x 64 bf16 = 512 x 16B chunks, 4 per thread
#pragma unroll
  for (int i = 0; i < 4; ++i) {
    const int idx = tid + i * 128;
    const int row = idx >> 3;
    const int c   = (idx & 7) * 8;
    *(uint4*)(Qs + row * A_LD + c) =
        *(const uint4*)(qbase + (size_t)(qt * A_BR + row) * rs + c);
  }

  v8f oacc[4];
#pragma unroll
  for (int n = 0; n < 4; ++n)
#pragma unroll
    for (int r = 0; r < 8; ++r) oacc[n][r] = 0.0f;
  float mrow[8], lrow[8];
#pragma unroll
  for (int r = 0; r < 8; ++r) { mrow[r] = NEG_INF; lrow[r] = 0.0f; }

  for (int kt = 0; kt <= qt; ++kt) {
    // K tile (raw copy) + V tile (transposed) into LDS
#pragma unroll
    for (int i = 0; i < 4; ++i) {
      const int idx = tid + i * 128;
      const int row = idx >> 3;
      const int c   = (idx & 7) * 8;
      *(uint4*)(Ks + row * A_LD + c) =
          *(const uint4*)(kbase + (size_t)(kt * A_BC + row) * rs + c);
      const uint4 vv = *(const uint4*)(vbase + (size_t)(kt * A_BC + row) * rs + c);
      const __bf16* ve = (const __bf16*)&vv;
#pragma unroll
      for (int j = 0; j < 8; ++j) Vt[(c + j) * A_LD + row] = ve[j];
    }
    __syncthreads();

    // S = Q K^T : 16 rows x 64 keys, Dh=64 -> 2 WMMA K-steps
    v8f sacc[4];
#pragma unroll
    for (int n = 0; n < 4; ++n)
#pragma unroll
      for (int r = 0; r < 8; ++r) sacc[n][r] = 0.0f;
#pragma unroll
    for (int ks = 0; ks < 2; ++ks) {
      const v16bf a = frag_a(Qs + (wave * 16 + l16) * A_LD, ks * 32 + half * 16);
#pragma unroll
      for (int n = 0; n < 4; ++n) {
        const v16bf bk = frag_b(Ks + (n * 16 + l16) * A_LD, ks * 32 + half * 16);
        sacc[n] = wmma_bf16(a, bk, sacc[n]);
      }
    }

    // scale + causal mask (diagonal tile only)
#pragma unroll
    for (int n = 0; n < 4; ++n)
#pragma unroll
      for (int r = 0; r < 8; ++r) sacc[n][r] *= 0.125f;
    if (kt == qt) {
#pragma unroll
      for (int n = 0; n < 4; ++n) {
        const int kcol = n * 16 + l16;
#pragma unroll
        for (int r = 0; r < 8; ++r) {
          const int qrow = wave * 16 + half * 8 + r;
          if (kcol > qrow) sacc[n][r] = NEG_INF;
        }
      }
    }

    // online softmax (rows live across the 16 lanes of each half)
#pragma unroll
    for (int r = 0; r < 8; ++r) {
      float t = fmaxf(fmaxf(sacc[0][r], sacc[1][r]),
                      fmaxf(sacc[2][r], sacc[3][r]));
#pragma unroll
      for (int off = 8; off >= 1; off >>= 1)
        t = fmaxf(t, __shfl_xor(t, off, 32));
      const float mnew = fmaxf(mrow[r], t);
      const float corr = __expf(mrow[r] - mnew);
      mrow[r] = mnew;
      float rsum = 0.0f;
#pragma unroll
      for (int n = 0; n < 4; ++n) {
        const float p = __expf(sacc[n][r] - mnew);
        sacc[n][r] = p;
        rsum += p;
      }
#pragma unroll
      for (int off = 8; off >= 1; off >>= 1)
        rsum += __shfl_xor(rsum, off, 32);
      lrow[r] = lrow[r] * corr + rsum;
#pragma unroll
      for (int n = 0; n < 4; ++n) oacc[n][r] *= corr;
    }

    // stage P as bf16 (wave-local rows; DS ops in-order within a wave)
#pragma unroll
    for (int n = 0; n < 4; ++n)
#pragma unroll
      for (int r = 0; r < 8; ++r)
        Ps[(wave * 16 + half * 8 + r) * A_LD + n * 16 + l16] = (__bf16)sacc[n][r];

    // O += P V : K-dim = 64 keys -> 2 WMMA K-steps
#pragma unroll
    for (int ks = 0; ks < 2; ++ks) {
      const v16bf a = frag_a(Ps + (wave * 16 + l16) * A_LD, ks * 32 + half * 16);
#pragma unroll
      for (int n = 0; n < 4; ++n) {
        const v16bf bv = frag_b(Vt + (n * 16 + l16) * A_LD, ks * 32 + half * 16);
        oacc[n] = wmma_bf16(a, bv, oacc[n]);
      }
    }
    __syncthreads();
  }

  // normalize + write merged-head bf16 output [B, S, D]
#pragma unroll
  for (int r = 0; r < 8; ++r) {
    const float inv = 1.0f / lrow[r];
    const int srow  = qt * A_BR + wave * 16 + half * 8 + r;
#pragma unroll
    for (int n = 0; n < 4; ++n) {
      const int col = h * A_DH + n * 16 + l16;
      out[((size_t)b * A_S + srow) * A_D + col] = (__bf16)(oacc[n][r] * inv);
    }
  }
}

// ---------------------------------------------------------------------------
// Launch pipeline
// ---------------------------------------------------------------------------
extern "C" void kernel_launch(void* const* d_in, const int* in_sizes, int n_in,
                              void* d_out, int out_size, void* d_ws, size_t ws_size,
                              hipStream_t stream) {
  const float* hidden   = (const float*)d_in[0];  // [4,2048,1024]
  const float* c_attn_w = (const float*)d_in[1];  // [1024,3072]
  const float* c_attn_b = (const float*)d_in[2];  // [3072]
  const float* c_proj_w = (const float*)d_in[3];  // [1024,1024]
  const float* c_proj_b = (const float*)d_in[4];  // [1024]
  float* out = (float*)d_out;                     // [4,2048,1024] fp32

  const int M = 4 * 2048;  // 8192 tokens

  // workspace (all bf16): 44M elements = 88 MB
  __bf16* hb   = (__bf16*)d_ws;                 // hidden        [8192,1024]
  __bf16* awt  = hb + (size_t)M * 1024;         // attn W^T      [3072,1024]
  __bf16* pwt  = awt + (size_t)3072 * 1024;     // proj W^T      [1024,1024]
  __bf16* qkvb = pwt + (size_t)1024 * 1024;     // qkv           [8192,3072]
  __bf16* attb = qkvb + (size_t)M * 3072;       // attn out      [8192,1024]

  // prepasses: convert activations, transpose+convert weights
  convert_f32_bf16<<<(unsigned)(((size_t)M * 1024 / 8 + 255) / 256), 256, 0, stream>>>(
      hidden, hb, (long)M * 1024);
  transpose_convert<<<dim3(3072 / 32, 1024 / 32), 256, 0, stream>>>(
      c_attn_w, awt, 1024, 3072);
  transpose_convert<<<dim3(1024 / 32, 1024 / 32), 256, 0, stream>>>(
      c_proj_w, pwt, 1024, 1024);

  // 1) fused QKV projection (bf16 out)
  gemm_bf16_bias<true><<<dim3(3072 / G_BN, M / G_BM), 256, 0, stream>>>(
      hb, awt, c_attn_b, qkvb, M, 3072, 1024);
  // 2) causal flash attention
  flash_attn_kernel<<<dim3(A_S / A_BR, 16, 4), 128, 0, stream>>>(qkvb, attb);
  // 3) output projection (fp32 out)
  gemm_bf16_bias<false><<<dim3(1024 / G_BN, M / G_BM), 256, 0, stream>>>(
      attb, pwt, c_proj_b, out, M, 1024, 1024);
}
